// GCN_79370995630475
// MI455X (gfx1250) — matrix operand order repaired and checked by reading, not compiled, not run on hardware
//
#include <hip/hip_runtime.h>
#include <hip/hip_bf16.h>

// ---------------------------------------------------------------------------
// GCN: 4 x (GEMM -> symmetric-normalized scatter-add -> bias [-> ReLU])
// GEMM path: bf16 WMMA (v_wmma_f32_16x16x32_bf16), fp32 accumulate.
//   - 256 threads / 8 waves per block, block tile 64x64, staged K = 64
//   - each wave: 16x32 output slab = 2 WMMAs sharing one A fragment,
//     unrolled over 2 K-steps -> 4 WMMAs per barrier pair
// Aggregation path: bandwidth-bound float4 gather + global_atomic_add_f32.
// ---------------------------------------------------------------------------

typedef __attribute__((ext_vector_type(16))) __bf16       v16bf;
typedef __attribute__((ext_vector_type(8)))  float        v8f;
typedef __attribute__((ext_vector_type(8)))  unsigned int v8u;

union BF16x16 { v8u u; v16bf b; };

__device__ __forceinline__ unsigned short f2bf(float f) {
  // round-to-nearest-even fp32 -> bf16
  unsigned int u = __float_as_uint(f);
  u += 0x7FFFu + ((u >> 16) & 1u);
  return (unsigned short)(u >> 16);
}

#define BM 64           // block tile rows
#define BN 64           // block tile cols
#define BK 64           // staged K per barrier (two 32-K WMMA steps)
#define LDK 68          // padded LDS row stride in shorts: 136B rows ->
                        // 8B-aligned b64 stores, conflict-free fragment reads

// H[M,N] = X[M,K] @ W[K,N].  K and N must be multiples of 64 (256/512 here).
__global__ __launch_bounds__(256) void gemm_wmma_bf16(
    const float* __restrict__ X, const float* __restrict__ W,
    float* __restrict__ H, int M, int K, int N)
{
  __shared__ __align__(16) unsigned short lA[BM][LDK]; // [row][k]  bf16
  __shared__ __align__(16) unsigned short lB[BN][LDK]; // [col][k]  bf16

  const int tid   = threadIdx.x;
  const int wave  = tid >> 5;
  const int lane  = tid & 31;
  const int wm    = (wave >> 1) * 16;   // wave row offset: 0/16/32/48
  const int wn    = (wave & 1)  * 32;   // wave col offset: 0/32 (two 16-wide tiles)
  const int bm    = blockIdx.y * BM;
  const int bn    = blockIdx.x * BN;
  const int l15   = lane & 15;
  const int khalf = lane >> 4;          // lane half selects K sub-range (ISA 7.12.2)

  v8f acc0 = {0.f,0.f,0.f,0.f,0.f,0.f,0.f,0.f};
  v8f acc1 = {0.f,0.f,0.f,0.f,0.f,0.f,0.f,0.f};

  for (int k0 = 0; k0 < K; k0 += BK) {
    // ---- stage A: 64 rows x 64 K, float4 global loads, b64 LDS stores ----
#pragma unroll
    for (int t = 0; t < 4; ++t) {
      int idx = tid + t * 256;          // 0..1023
      int r   = idx >> 4;               // row within tile
      int k4  = (idx & 15) << 2;        // k offset (multiple of 4)
      int gr  = bm + r;
      float4 v = {0.f, 0.f, 0.f, 0.f};
      if (gr < M) v = *(const float4*)(X + gr * K + k0 + k4);
      unsigned int p0 = ((unsigned int)f2bf(v.y) << 16) | f2bf(v.x);
      unsigned int p1 = ((unsigned int)f2bf(v.w) << 16) | f2bf(v.z);
      *(unsigned long long*)&lA[r][k4] =
          ((unsigned long long)p1 << 32) | (unsigned long long)p0;
    }
    // ---- stage B: 64 K x 64 N, coalesced float4 along N, col-major LDS ----
#pragma unroll
    for (int t = 0; t < 4; ++t) {
      int idx = tid + t * 256;
      int kk  = idx >> 4;
      int c4  = (idx & 15) << 2;
      float4 v = *(const float4*)(W + (k0 + kk) * N + bn + c4);
      lB[c4 + 0][kk] = f2bf(v.x);
      lB[c4 + 1][kk] = f2bf(v.y);
      lB[c4 + 2][kk] = f2bf(v.z);
      lB[c4 + 3][kk] = f2bf(v.w);
    }
    __syncthreads();

    // ---- 2 K-steps x 2 N-tiles = 4 WMMAs per barrier pair ----
#pragma unroll
    for (int ks = 0; ks < BK; ks += 32) {
      BF16x16 a, b0, b1;
#pragma unroll
      for (int v = 0; v < 8; ++v) {
        // A 16x32: lanes 0-15 -> K 0-15 (2/VGPR); lanes 16-31 -> K 16-31
        int ka = ks + khalf * 16 + v * 2;
        a.u[v]  = *(const unsigned int*)&lA[wm + l15][ka];
        // B 32x16: lanes 0-15 -> K 0-15 (2/VGPR); lanes 16-31 -> K 16-31
        int kb = ks + khalf * 16 + v * 2;
        b0.u[v] = *(const unsigned int*)&lB[wn + l15][kb];
        b1.u[v] = *(const unsigned int*)&lB[wn + 16 + l15][kb];
      }
      acc0 = __builtin_amdgcn_wmma_f32_16x16x32_bf16(
          false, a.b, false, b0.b, (short)0, acc0, false, false);
      acc1 = __builtin_amdgcn_wmma_f32_16x16x32_bf16(
          false, a.b, false, b1.b, (short)0, acc1, false, false);
    }
    __syncthreads();
  }

  // D layout: VGPR r -> M = r (+8 for upper lane half); N = lane&15
  int row0 = bm + wm + khalf * 8;
  int col0 = bn + wn + l15;
  if (bm + BM <= M) {
#pragma unroll
    for (int r = 0; r < 8; ++r) {
      H[(row0 + r) * N + col0]      = acc0[r];
      H[(row0 + r) * N + col0 + 16] = acc1[r];
    }
  } else {
#pragma unroll
    for (int r = 0; r < 8; ++r) {
      if (row0 + r < M) {
        H[(row0 + r) * N + col0]      = acc0[r];
        H[(row0 + r) * N + col0 + 16] = acc1[r];
      }
    }
  }
}

// ---------------------------------------------------------------------------

__global__ void fill_kernel(float* __restrict__ p, float val, int n) {
  int i = blockIdx.x * blockDim.x + threadIdx.x;
  if (i < n) p[i] = val;
}

__global__ void deg_edges_kernel(const int* __restrict__ col,
                                 float* __restrict__ deg, int E) {
  int i = blockIdx.x * blockDim.x + threadIdx.x;
  if (i < E) atomicAdd(&deg[col[i]], 1.0f);
}

__global__ void deg_to_dis_kernel(float* __restrict__ d, int n) {
  int i = blockIdx.x * blockDim.x + threadIdx.x;
  if (i < n) d[i] = rsqrtf(d[i]);  // deg >= 1 always (self loop pre-counted)
}

// One thread per (edge, 4-feature chunk). e in [E, E+Nn) are the self loops.
__global__ void agg_kernel(const float* __restrict__ H,
                           const int* __restrict__ rows,
                           const int* __restrict__ cols,
                           const float* __restrict__ dis,
                           float* __restrict__ out,
                           int E, int Nn, int C)
{
  int idx = blockIdx.x * blockDim.x + threadIdx.x;
  int chunks = C >> 2;               // C/4 float4 chunks
  int e = idx / chunks;
  if (e >= E + Nn) return;
  int f = (idx - e * chunks) << 2;
  int r, c;
  if (e < E) { r = rows[e]; c = cols[e]; }
  else       { r = c = e - E; }
  float nrm = dis[r] * dis[c];
  const float4 h = *(const float4*)(H + r * C + f);
  float* o = out + c * C + f;
  atomicAdd(o + 0, h.x * nrm);
  atomicAdd(o + 1, h.y * nrm);
  atomicAdd(o + 2, h.z * nrm);
  atomicAdd(o + 3, h.w * nrm);
}

__global__ void bias_act_kernel(float* __restrict__ buf,
                                const float* __restrict__ b,
                                int total, int Cmask, int relu) {
  int i = blockIdx.x * blockDim.x + threadIdx.x;
  if (i >= total) return;
  float v = buf[i] + b[i & Cmask];   // C = 512, power of two
  buf[i] = relu ? fmaxf(v, 0.0f) : v;
}

// ---------------------------------------------------------------------------

extern "C" void kernel_launch(void* const* d_in, const int* in_sizes, int n_in,
                              void* d_out, int out_size, void* d_ws, size_t ws_size,
                              hipStream_t stream) {
  const float* x  = (const float*)d_in[0];
  const int*   ei = (const int*)  d_in[1];   // [2, E] flattened: rows then cols
  const float* W1 = (const float*)d_in[2];
  const float* b1 = (const float*)d_in[3];
  const float* W2 = (const float*)d_in[4];
  const float* b2 = (const float*)d_in[5];
  const float* W3 = (const float*)d_in[6];
  const float* b3 = (const float*)d_in[7];
  const float* W4 = (const float*)d_in[8];
  const float* b4 = (const float*)d_in[9];

  const int Cin = 256, C = 512;
  const int Nn  = in_sizes[0] / Cin;     // 10000 nodes
  const int E   = in_sizes[1] / 2;       // 160000 edges
  const int total = Nn * C;              // 5,120,000 floats

  const int* rows = ei;
  const int* cols = ei + E;

  // Workspace layout: dis | bufA (gemm out) | bufB (agg out / next input)
  float* dis  = (float*)d_ws;
  float* bufA = (float*)((char*)d_ws + (1 << 16));
  float* bufB = (float*)((char*)d_ws + (1 << 16) + (size_t)total * sizeof(float));
  float* outp = (float*)d_out;

  const int TPB = 256;
  dim3 gGemm(C / BN, (Nn + BM - 1) / BM);
  int  gN     = (Nn + TPB - 1) / TPB;
  int  gE     = (E + TPB - 1) / TPB;
  int  gTot   = (total + TPB - 1) / TPB;
  int  aggW   = (E + Nn) * (C >> 2);
  int  gAgg   = (aggW + TPB - 1) / TPB;

  // --- degree -> dis = deg^-1/2 (self-loop counted via init to 1.0) ---
  fill_kernel<<<gN, TPB, 0, stream>>>(dis, 1.0f, Nn);
  deg_edges_kernel<<<gE, TPB, 0, stream>>>(cols, dis, E);
  deg_to_dis_kernel<<<gN, TPB, 0, stream>>>(dis, Nn);

  // --- layer 1 ---
  gemm_wmma_bf16<<<gGemm, TPB, 0, stream>>>(x, W1, bufA, Nn, Cin, C);
  fill_kernel<<<gTot, TPB, 0, stream>>>(bufB, 0.0f, total);
  agg_kernel<<<gAgg, TPB, 0, stream>>>(bufA, rows, cols, dis, bufB, E, Nn, C);
  bias_act_kernel<<<gTot, TPB, 0, stream>>>(bufB, b1, total, C - 1, 1);

  // --- layer 2 ---
  gemm_wmma_bf16<<<gGemm, TPB, 0, stream>>>(bufB, W2, bufA, Nn, C, C);
  fill_kernel<<<gTot, TPB, 0, stream>>>(bufB, 0.0f, total);
  agg_kernel<<<gAgg, TPB, 0, stream>>>(bufA, rows, cols, dis, bufB, E, Nn, C);
  bias_act_kernel<<<gTot, TPB, 0, stream>>>(bufB, b2, total, C - 1, 1);

  // --- layer 3 ---
  gemm_wmma_bf16<<<gGemm, TPB, 0, stream>>>(bufB, W3, bufA, Nn, C, C);
  fill_kernel<<<gTot, TPB, 0, stream>>>(bufB, 0.0f, total);
  agg_kernel<<<gAgg, TPB, 0, stream>>>(bufA, rows, cols, dis, bufB, E, Nn, C);
  bias_act_kernel<<<gTot, TPB, 0, stream>>>(bufB, b3, total, C - 1, 1);

  // --- layer 4 (no ReLU, aggregate straight into d_out) ---
  gemm_wmma_bf16<<<gGemm, TPB, 0, stream>>>(bufB, W4, bufA, Nn, C, C);
  fill_kernel<<<gTot, TPB, 0, stream>>>(outp, 0.0f, total);
  agg_kernel<<<gAgg, TPB, 0, stream>>>(bufA, rows, cols, dis, outp, E, Nn, C);
  bias_act_kernel<<<gTot, TPB, 0, stream>>>(outp, b4, total, C - 1, 0);
}
